// VisionMoEAdapter_41334765257023
// MI455X (gfx1250) — matrix operand order repaired and checked by reading, not compiled, not run on hardware
//
#include <hip/hip_runtime.h>
#include <hip/hip_bf16.h>

// ---------------- problem constants ----------------
constexpr int kN = 32768;   // tokens
constexpr int kD = 1024;    // model dim
constexpr int kE = 8;       // experts
constexpr int kH = 4096;    // expert hidden
constexpr int kBM = 16;     // token rows per block in fused kernel
constexpr int kHC = 128;    // H-chunk width
constexpr float kEps = 1e-5f;

typedef unsigned short u16;
typedef __attribute__((ext_vector_type(16))) __bf16 v16bf;
typedef __attribute__((ext_vector_type(8)))  float  v8f;

// fp32 -> bf16 bits, round-to-nearest-even
__device__ __forceinline__ u16 f2bf(float x) {
  unsigned int u = __float_as_uint(x);
  u += 0x7FFFu + ((u >> 16) & 1u);
  return (u16)(u >> 16);
}

union FragU { v16bf v; uint4 u4[2]; };

// A fragment (16x32 bf16) from an LDS tile, row-major with given stride.
// ISA 7.12.2: lane m=lane&15; lanes<16 hold K={k..k+7, k+16..k+23},
// lanes>=16 hold K={k+8..k+15, k+24..k+31}  -> two contiguous 16B chunks.
__device__ __forceinline__ v16bf load_a_lds(const u16* base, int stride, int k, int lane) {
  const int m = lane & 15, hi = lane >> 4;
  const u16* p = base + m * stride + k + hi * 8;
  FragU f;
  f.u4[0] = *(const uint4*)(p);
  f.u4[1] = *(const uint4*)(p + 16);
  return f.v;
}

// B fragment (32x16 bf16) from global, weights pre-transposed so that the
// 16 K-values a lane needs (column n, K = {k+hi*8..+7, k+16+hi*8..+7})
// are two 16B chunks: p points at &Wt[col][k + hi*8].
__device__ __forceinline__ v16bf load_b_gl(const u16* p) {
  FragU f;
  f.u4[0] = *(const uint4*)(p);
  f.u4[1] = *(const uint4*)(p + 16);
  return f.v;
}

__device__ __forceinline__ v8f wmma_bf16(v16bf a, v16bf b, v8f c) {
  return __builtin_amdgcn_wmma_f32_16x16x32_bf16(
      false, a, false, b, (short)0, c, false, false);
}

// fast SiLU: result feeds bf16 (8 mantissa bits), so HW v_rcp_f32 is plenty
__device__ __forceinline__ float fast_silu(float v) {
  return v * __builtin_amdgcn_rcpf(1.f + __expf(-v));
}

// ---------------- kernel 0: zero accumulator + counters ----------------
__global__ __launch_bounds__(256)
void moe_zero_kernel(float* __restrict__ out_acc, int* __restrict__ counts) {
  const unsigned total4 = (unsigned)(kN * (kD / 4));
  unsigned i = blockIdx.x * blockDim.x + threadIdx.x;
  const unsigned stride = gridDim.x * blockDim.x;
  float4 z = make_float4(0.f, 0.f, 0.f, 0.f);
  float4* p = (float4*)out_acc;
  for (unsigned j = i; j < total4; j += stride) p[j] = z;
  if (blockIdx.x == 0 && threadIdx.x < kE) counts[threadIdx.x] = 0;
}

// ---------------- kernel 1: x fp32 -> bf16 ----------------
__global__ __launch_bounds__(256)
void moe_cvt_x_kernel(const float* __restrict__ x, u16* __restrict__ xb) {
  const unsigned total = (unsigned)(kN * kD);
  unsigned i = blockIdx.x * blockDim.x + threadIdx.x;
  const unsigned stride = gridDim.x * blockDim.x;
  for (unsigned j = i; j < total; j += stride) xb[j] = f2bf(x[j]);
}

// W1 [E][D][H] fp32 -> W1t [E][H][D] bf16 (so GEMM1 B-frags are contiguous in K=D)
__global__ __launch_bounds__(256)
void moe_tr_w1_kernel(const float* __restrict__ W1, u16* __restrict__ W1t) {
  const unsigned total = (unsigned)kE * kD * kH;
  unsigned i = blockIdx.x * blockDim.x + threadIdx.x;
  const unsigned stride = gridDim.x * blockDim.x;
  for (unsigned j = i; j < total; j += stride) {
    unsigned d = j % kD, h = (j / kD) % kH, e = j / (kD * kH);
    W1t[j] = f2bf(W1[((size_t)e * kD + d) * kH + h]);
  }
}

// W2 [E][H][D] fp32 -> W2t [E][D][H] bf16 (GEMM2 B-frags contiguous in K=H)
__global__ __launch_bounds__(256)
void moe_tr_w2_kernel(const float* __restrict__ W2, u16* __restrict__ W2t) {
  const unsigned total = (unsigned)kE * kD * kH;
  unsigned i = blockIdx.x * blockDim.x + threadIdx.x;
  const unsigned stride = gridDim.x * blockDim.x;
  for (unsigned j = i; j < total; j += stride) {
    unsigned k = j % kH, n = (j / kH) % kD, e = j / (kD * kH);
    W2t[j] = f2bf(W2[((size_t)e * kH + k) * kD + n]);
  }
}

// ---------------- kernel 2: router + top-2 scatter ----------------
__global__ __launch_bounds__(256)
void moe_router_kernel(const float* __restrict__ x, const float* __restrict__ Wr,
                       const float* __restrict__ br, int* __restrict__ counts,
                       int* __restrict__ tok_idx, float* __restrict__ tok_w) {
  __shared__ float WrS[kD * kE];   // 32 KB
  for (int i = threadIdx.x; i < kD * kE; i += 256) WrS[i] = Wr[i];
  __syncthreads();

  const int n = blockIdx.x * 256 + threadIdx.x;
  float acc[kE];
#pragma unroll
  for (int e = 0; e < kE; ++e) acc[e] = br[e];
  const float4* xr = (const float4*)(x + (size_t)n * kD);
  for (int d4 = 0; d4 < kD / 4; ++d4) {
    float4 xv = xr[d4];
    const float* w = &WrS[d4 * 4 * kE];
#pragma unroll
    for (int e = 0; e < kE; ++e)
      acc[e] += xv.x * w[e] + xv.y * w[kE + e] + xv.z * w[2 * kE + e] + xv.w * w[3 * kE + e];
  }
  // top-2 (first-occurrence tie-break like jax top_k)
  int e0 = 0; float l0 = acc[0];
#pragma unroll
  for (int e = 1; e < kE; ++e) if (acc[e] > l0) { l0 = acc[e]; e0 = e; }
  int e1 = (e0 == 0) ? 1 : 0; float l1 = acc[e1];
#pragma unroll
  for (int e = 0; e < kE; ++e)
    if (e != e0 && acc[e] > l1) { l1 = acc[e]; e1 = e; }
  // renormalized top-2 softmax weights
  const float w0 = 1.f / (1.f + __expf(l1 - l0));
  const float w1 = 1.f - w0;
  int p0 = atomicAdd(&counts[e0], 1);
  tok_idx[e0 * kN + p0] = n; tok_w[e0 * kN + p0] = w0;
  int p1 = atomicAdd(&counts[e1], 1);
  tok_idx[e1 * kN + p1] = n; tok_w[e1 * kN + p1] = w1;
}

// ---------------- kernel 3: fused expert FFN (WMMA bf16) ----------------
// grid = (ceil(N/BM), E), block = 256 (8 waves). Each block: 16 gathered
// tokens of one expert; loops H in 128-chunks; GEMM1 -> SiLU -> LDS,
// GEMM2 accumulates full 16x1024 output in registers (8 tiles/wave).
__global__ __launch_bounds__(256)
void moe_fused_kernel(const u16* __restrict__ xb, const u16* __restrict__ W1t,
                      const u16* __restrict__ W2t, const float* __restrict__ b1,
                      const float* __restrict__ b2, const int* __restrict__ counts,
                      const int* __restrict__ tok_idx, const float* __restrict__ tok_w,
                      float* __restrict__ out_acc) {
  const int e = blockIdx.y;
  const int cnt = counts[e];
  const int m0 = blockIdx.x * kBM;
  if (m0 >= cnt) return;

  __shared__ u16  Xs[kBM * kD];    // 32 KB gathered token rows (bf16)
  __shared__ u16  Hs[kBM * kHC];   // 4 KB  current hidden chunk (bf16)
  __shared__ int   toks[kBM];
  __shared__ float wts[kBM];

  const int tid = threadIdx.x;
  const int wave = tid >> 5;          // 0..7
  const int lane = tid & 31;

  if (tid < kBM) {
    int m = m0 + tid;
    int tok = 0; float w = 0.f;
    if (m < cnt) { tok = tok_idx[e * kN + m]; w = tok_w[e * kN + m]; }
    toks[tid] = tok; wts[tid] = w;    // w==0 rows contribute nothing
  }
  __syncthreads();

  { // gather 16 rows of bf16 x into LDS (16B vector loads, coalesced)
    const int r = tid >> 4;
    const int c0 = (tid & 15) * 64;
    const uint4* src = (const uint4*)(xb + (size_t)toks[r] * kD + c0);
    uint4* dst = (uint4*)(Xs + r * kD + c0);
#pragma unroll
    for (int i = 0; i < 8; ++i) dst[i] = src[i];
  }
  __syncthreads();

  v8f acc2[8];
#pragma unroll
  for (int t = 0; t < 8; ++t) acc2[t] = {};

  const int nloc = lane & 15;
  const int hi   = lane >> 4;

  for (int hc = 0; hc < kH; hc += kHC) {
    // ---- GEMM1: this wave computes 16x16 tile at columns [hcol, hcol+16)
    const int hcol = hc + wave * 16;
    const u16* w1p = W1t + ((size_t)e * kH + hcol + nloc) * kD + hi * 8;
    v8f acc1 = {};
#pragma unroll 4
    for (int k = 0; k < kD; k += 32) {
      __builtin_prefetch(w1p + k + 256, 0, 1);           // global_prefetch_b8
      v16bf a = load_a_lds(Xs, kD, k, lane);
      v16bf b = load_b_gl(w1p + k);
      acc1 = wmma_bf16(a, b, acc1);
    }
    { // epilogue: +b1, SiLU, bf16 -> Hs
      const float bb = b1[e * kH + hcol + nloc];
      const int mrow0 = hi * 8;
#pragma unroll
      for (int r = 0; r < 8; ++r) {
        Hs[(mrow0 + r) * kHC + wave * 16 + nloc] = f2bf(fast_silu(acc1[r] + bb));
      }
    }
    __syncthreads();

    // ---- GEMM2: wave owns output cols [wave*128, wave*128+128)
#pragma unroll
    for (int k = 0; k < kHC; k += 32) {
      v16bf a = load_a_lds(Hs, kHC, k, lane);
#pragma unroll
      for (int t = 0; t < 8; ++t) {
        const int col = wave * 128 + t * 16 + nloc;
        const u16* w2p = W2t + ((size_t)e * kD + col) * kH + hc + k + hi * 8;
        v16bf b = load_b_gl(w2p);
        acc2[t] = wmma_bf16(a, b, acc2[t]);
      }
    }
    __syncthreads();
  }

  // ---- epilogue: +b2, scale by routing weight, atomic combine (K=2 writers)
  const int mrow0 = hi * 8;
#pragma unroll
  for (int t = 0; t < 8; ++t) {
    const int col = wave * 128 + t * 16 + nloc;
    const float bb = b2[e * kD + col];
#pragma unroll
    for (int r = 0; r < 8; ++r) {
      const int m = mrow0 + r;
      const float val = (acc2[t][r] + bb) * wts[m];
      atomicAdd(&out_acc[(size_t)toks[m] * kD + col], val);
    }
  }
}

// ---------------- kernel 4: residual gate + LayerNorm ----------------
__global__ __launch_bounds__(256)
void moe_ln_kernel(const float* __restrict__ x, const float* __restrict__ out_acc,
                   const float* __restrict__ gate, const float* __restrict__ gamma,
                   const float* __restrict__ beta, float* __restrict__ out) {
  const int n = blockIdx.x, t = threadIdx.x;
  const float g = gate[0];
  float4 xv = ((const float4*)(x + (size_t)n * kD))[t];
  float4 av = ((const float4*)(out_acc + (size_t)n * kD))[t];
  float4 z;
  z.x = xv.x + g * av.x; z.y = xv.y + g * av.y;
  z.z = xv.z + g * av.z; z.w = xv.w + g * av.w;

  __shared__ float r1[256], r2[256];
  r1[t] = z.x + z.y + z.z + z.w;
  r2[t] = z.x * z.x + z.y * z.y + z.z * z.z + z.w * z.w;
  __syncthreads();
  for (int o = 128; o > 0; o >>= 1) {
    if (t < o) { r1[t] += r1[t + o]; r2[t] += r2[t + o]; }
    __syncthreads();
  }
  const float mu = r1[0] * (1.f / kD);
  const float var = r2[0] * (1.f / kD) - mu * mu;
  const float rstd = rsqrtf(var + kEps);

  float4 gm = ((const float4*)gamma)[t];
  float4 bt = ((const float4*)beta)[t];
  float4 o4;
  o4.x = (z.x - mu) * rstd * gm.x + bt.x;
  o4.y = (z.y - mu) * rstd * gm.y + bt.y;
  o4.z = (z.z - mu) * rstd * gm.z + bt.z;
  o4.w = (z.w - mu) * rstd * gm.w + bt.w;
  ((float4*)(out + (size_t)n * kD))[t] = o4;
}

// ---------------- launcher ----------------
extern "C" void kernel_launch(void* const* d_in, const int* in_sizes, int n_in,
                              void* d_out, int out_size, void* d_ws, size_t ws_size,
                              hipStream_t stream) {
  const float* x     = (const float*)d_in[0];
  const float* Wr    = (const float*)d_in[1];
  const float* br    = (const float*)d_in[2];
  const float* W1    = (const float*)d_in[3];
  const float* b1    = (const float*)d_in[4];
  const float* W2    = (const float*)d_in[5];
  const float* b2    = (const float*)d_in[6];
  const float* gate  = (const float*)d_in[7];
  const float* gamma = (const float*)d_in[8];
  const float* beta  = (const float*)d_in[9];
  float* out = (float*)d_out;

  // workspace layout (all 16B-aligned)
  char* ws = (char*)d_ws;
  size_t off = 0;
  float* out_acc = (float*)(ws + off); off += (size_t)kN * kD * 4;       // 128 MiB
  u16*   xb      = (u16*)(ws + off);  off += (size_t)kN * kD * 2;        //  64 MiB
  u16*   W1t     = (u16*)(ws + off);  off += (size_t)kE * kH * kD * 2;   //  64 MiB
  u16*   W2t     = (u16*)(ws + off);  off += (size_t)kE * kD * kH * 2;   //  64 MiB
  int*   counts  = (int*)(ws + off);  off += 256;
  int*   tok_idx = (int*)(ws + off);  off += (size_t)kE * kN * 4;        //   1 MiB
  float* tok_w   = (float*)(ws + off); off += (size_t)kE * kN * 4;       //   1 MiB

  moe_zero_kernel<<<4096, 256, 0, stream>>>(out_acc, counts);
  moe_cvt_x_kernel<<<4096, 256, 0, stream>>>(x, xb);
  moe_tr_w1_kernel<<<8192, 256, 0, stream>>>(W1, W1t);
  moe_tr_w2_kernel<<<8192, 256, 0, stream>>>(W2, W2t);
  moe_router_kernel<<<kN / 256, 256, 0, stream>>>(x, Wr, br, counts, tok_idx, tok_w);

  dim3 grid((kN + kBM - 1) / kBM, kE);
  moe_fused_kernel<<<grid, 256, 0, stream>>>(xb, W1t, W2t, b1, b2, counts,
                                             tok_idx, tok_w, out_acc);
  moe_ln_kernel<<<kN, 256, 0, stream>>>(x, out_acc, gate, gamma, beta, out);
}
